// PIR_NCDE_84602265797029
// MI455X (gfx1250) — matrix operand order repaired and checked by reading, not compiled, not run on hardware
//
#include <hip/hip_runtime.h>
#include <hip/hip_bf16.h>
#include <math.h>

// ---------------------------------------------------------------------------
// CDNA5 (gfx1250) wave32 WMMA implementation of the neural-CDE + transformer
// reference. All heavy GEMMs use v_wmma_f32_16x16x32_f16 (f32 accumulate).
// Activations are cached in LDS as row-major f16 so every WMMA fragment is
// fed by two ds_load_b128 (contiguous k = 16*(lane/16) + s order, same in A and B),
// with fp32->f16 conversion done once at produce time.
// ---------------------------------------------------------------------------

typedef __attribute__((ext_vector_type(16))) _Float16 v16h;
typedef __attribute__((ext_vector_type(8)))  float    v8f;

__device__ __forceinline__ v8f wmma32(v16h a, v16h b, v8f c) {
  return __builtin_amdgcn_wmma_f32_16x16x32_f16(false, a, false, b, (short)0, c,
                                                false, false);
}

__device__ __forceinline__ unsigned pkh2(float a, float b) {
  union { _Float16 h[2]; unsigned u; } p;
  p.h[0] = (_Float16)a;
  p.h[1] = (_Float16)b;
  return p.u;
}

// Load 8 dwords (one pre-packed f16 B-fragment slice for this lane) with two
// b128 global loads.
__device__ __forceinline__ v16h load8u(const unsigned* p) {
  union { v16h v; uint4 q[2]; } r;
  r.q[0] = *(const uint4*)(p);
  r.q[1] = *(const uint4*)(p + 4);
  return r.v;
}

// A fragment (16x32, f16) from a row-major f16 LDS tile (ldkHalf halves/row).
// Lane holds row M=lane&15; halves are 16 consecutive at
// K = k0 + (lane>>4)*16  ->  2x ds_load_b128.
__device__ __forceinline__ v16h packA_h16(const unsigned* baseU, int ldkHalf,
                                          int k0, int lane) {
  union { v16h v; uint4 q[2]; } r;
  int row = lane & 15;
  int hw = (lane >> 4) << 4;
  const uint4* p = (const uint4*)(baseU + (((row * ldkHalf) + k0 + hw) >> 1));
  r.q[0] = p[0];
  r.q[1] = p[1];
  return r.v;
}

// B fragment (32x16, f16) from a row-major f16 LDS tile holding W[n][k].
// Lane holds column N=lane&15; halves are 16 consecutive at
// K = k0 + (lane>>4)*16  ->  2x ds_load_b128.
__device__ __forceinline__ v16h packB_h16(const unsigned* baseU, int ldkHalf,
                                          int k0, int lane) {
  union { v16h v; uint4 q[2]; } r;
  int col = lane & 15;
  int hw = (lane >> 4) << 4;
  const uint4* p = (const uint4*)(baseU + (((col * ldkHalf) + k0 + hw) >> 1));
  r.q[0] = p[0];
  r.q[1] = p[1];
  return r.v;
}

// ---------------------------------------------------------------------------
// Weight pre-pack: W (N x K fp32, row-major) -> f16 B-fragments in lane order:
//   out[((tile*KF + kf)*32 + lane)*8 + j]
// so each lane fetches its fragment with two b128 loads at use time.
// ---------------------------------------------------------------------------
__global__ void cde_pack_kernel(const float* __restrict__ W,
                                unsigned* __restrict__ out, int K, int tiles) {
  int id = blockIdx.x * 256 + threadIdx.x;
  int KF = K >> 5;
  int total = tiles * KF * 32 * 8;
  if (id >= total) return;
  int j = id & 7;
  int lane = (id >> 3) & 31;
  int rem = id >> 8;
  int kf = rem % KF;
  int tile = rem / KF;
  int n = tile * 16 + (lane & 15);
  int k = kf * 32 + ((lane >> 4) << 4) + (j << 1);
  out[id] = pkh2(W[(long long)n * K + k], W[(long long)n * K + k + 1]);
}

// ---------------------------------------------------------------------------
// Spline eval + initial projection:  z0[b,h] -> zT[b,0,:]
// sum_l X0 collapses to sum_{l<255} ca[l] + (ca+cb+c2/2+c3/3)[254].
// ---------------------------------------------------------------------------
__global__ void spline_z0_kernel(const float* __restrict__ coeffs,
                                 const float* __restrict__ initW,
                                 const float* __restrict__ initb,
                                 float* __restrict__ zT) {
  int b = blockIdx.x;
  int t = threadIdx.x;  // 0..127
  __shared__ float part[128];
  __shared__ float Ssm[16];
  int c = t & 15, chunk = t >> 4;
  float p = 0.0f;
  int l0 = chunk * 32;
  int l1 = l0 + 32; if (l1 > 255) l1 = 255;
  for (int l = l0; l < l1; ++l)
    p += coeffs[((long long)(b * 255 + l)) * 64 + c];
  part[t] = p;
  __syncthreads();
  if (t < 16) {
    float s = 0.0f;
    for (int ch = 0; ch < 8; ++ch) s += part[ch * 16 + t];
    const float* c254 = coeffs + ((long long)(b * 255 + 254)) * 64;
    s += c254[t] + c254[16 + t] + 0.5f * c254[32 + t] + c254[48 + t] * (1.0f / 3.0f);
    Ssm[t] = s;
  }
  __syncthreads();
  float z = 256.0f * initb[t];
#pragma unroll
  for (int cc = 0; cc < 16; ++cc) z += Ssm[cc] * initW[t * 16 + cc];
  zT[((long long)b * 256) * 128 + t] = z;
}

// ---------------------------------------------------------------------------
// Persistent RK4 integrator. 16 blocks x 256 threads (8 waves). Each block
// owns 16 batch rows (= WMMA M). Activations live in LDS as f16 (one cvt at
// produce time; every consume is 2x ds_load_b128 per fragment). Weights
// stream from pre-packed, L2-resident fragment buffers via b128 loads.
// ---------------------------------------------------------------------------
__global__ __launch_bounds__(256) void cde_integrate_kernel(
    const float* __restrict__ coeffs, const unsigned* __restrict__ pW0,
    const unsigned* __restrict__ pWm, const unsigned* __restrict__ pW4,
    const float* __restrict__ b0g, const float* __restrict__ bmg,
    const float* __restrict__ b4g, float* __restrict__ zT) {
  __shared__ float zbase[2048], accl[2048], garr[2048];
  __shared__ unsigned zargU[1024];   // 16 x 128 f16 (packed pairs)
  __shared__ unsigned hsmU[512];     // 16 x 64  f16
  __shared__ float dxA[256], dxB[256], dxC[256];
  _Float16* hsmH = (_Float16*)hsmU;

  int tid = threadIdx.x;
  int w = tid >> 5, lane = tid & 31;
  int bq = blockIdx.x * 16;

  // load z0 (pairs -> f32 state + packed f16 zarg)
  for (int e2 = tid; e2 < 1024; e2 += 256) {
    int e = e2 * 2;
    float2 z = *(const float2*)&zT[((long long)(bq + (e >> 7)) * 256) * 128 + (e & 127)];
    zbase[e] = z.x;
    zbase[e + 1] = z.y;
    zargU[e2] = pkh2(z.x, z.y);
  }

  // Preload W0 / Wm fragments into VGPRs (waves 0..3 own N-tiles 0..3).
  v16h w0f[4] = {};
  v16h wmf[2][2] = {};
  if (w < 4) {
#pragma unroll
    for (int kf = 0; kf < 4; ++kf)
      w0f[kf] = load8u(pW0 + ((w * 4 + kf) * 32 + lane) * 8);
#pragma unroll
    for (int layer = 0; layer < 2; ++layer)
#pragma unroll
      for (int kf = 0; kf < 2; ++kf)
        wmf[layer][kf] = load8u(pWm + (((layer * 4 + w) * 2 + kf) * 32 + lane) * 8);
  }
  __syncthreads();

  for (int n = 0; n < 255; ++n) {
    // dX/dt at t=n (k1), t=n+0.5 (k2,k3), t=n+1 (k4)
    for (int e = tid; e < 256; e += 256) {
      int bi = e >> 4, c = e & 15;
      const float* cc = coeffs + ((long long)((bq + bi) * 255 + n)) * 64;
      float cb = cc[16 + c], c2 = cc[32 + c], c3 = cc[48 + c];
      dxA[e] = cb;
      dxB[e] = cb + (c2 + 0.5f * c3) * 0.5f;
      dxC[e] = (n < 254) ? cc[64 + 16 + c] : (cb + c2 + c3);
    }
    __syncthreads();

    for (int s = 0; s < 4; ++s) {
      const float* dx = (s == 0) ? dxA : ((s == 3) ? dxC : dxB);

      // layer 0: h = relu(z @ W0^T + b0)   [16x128]x[128x64]
      if (w < 4) {
        v8f acc = {};
#pragma unroll
        for (int kf = 0; kf < 4; ++kf) {
          v16h a = packA_h16(zargU, 128, kf * 32, lane);
          acc = wmma32(a, w0f[kf], acc);
        }
        int nn = w * 16 + (lane & 15);
        float bv = b0g[nn];
#pragma unroll
        for (int r = 0; r < 8; ++r) {
          float v = acc[r] + bv;
          v = v > 0.0f ? v : 0.0f;
          hsmH[(r + ((lane >> 4) << 3)) * 64 + nn] = (_Float16)v;
        }
      }
      __syncthreads();

      // mid layers: h = relu(h @ Wm^T + bm)  [16x64]x[64x64]
      for (int layer = 0; layer < 2; ++layer) {
        v16h a0 = {}, a1 = {};
        if (w < 4) {
          a0 = packA_h16(hsmU, 64, 0, lane);
          a1 = packA_h16(hsmU, 64, 32, lane);
        }
        __syncthreads();
        if (w < 4) {
          v8f acc = {};
          acc = wmma32(a0, wmf[layer][0], acc);
          acc = wmma32(a1, wmf[layer][1], acc);
          int nn = w * 16 + (lane & 15);
          float bv = bmg[layer * 64 + nn];
#pragma unroll
          for (int r = 0; r < 8; ++r) {
            float v = acc[r] + bv;
            v = v > 0.0f ? v : 0.0f;
            hsmH[(r + ((lane >> 4) << 3)) * 64 + nn] = (_Float16)v;
          }
        }
        __syncthreads();
      }

      // wide layer + fused tanh + dX contraction:
      // out[16x2048] tile t covers (h_idx = t, c = 0..15);
      // g[b, t] = sum_c tanh(out)[b, c] * dx[b, c]
      {
        v16h a0 = packA_h16(hsmU, 64, 0, lane);
        v16h a1 = packA_h16(hsmU, 64, 32, lane);
        int cc = lane & 15;
        for (int t4 = w; t4 < 128; t4 += 8) {
          v8f acc = {};
          v16h bf0 = load8u(pW4 + ((t4 * 2 + 0) * 32 + lane) * 8);
          v16h bf1 = load8u(pW4 + ((t4 * 2 + 1) * 32 + lane) * 8);
          acc = wmma32(a0, bf0, acc);
          acc = wmma32(a1, bf1, acc);
          float bvb = b4g[t4 * 16 + cc];
#pragma unroll
          for (int r = 0; r < 8; ++r) {
            int M = r + ((lane >> 4) << 3);
            float v = tanhf(acc[r] + bvb);
            float p = v * dx[M * 16 + cc];
            p += __shfl_xor(p, 1, 32);
            p += __shfl_xor(p, 2, 32);
            p += __shfl_xor(p, 4, 32);
            p += __shfl_xor(p, 8, 32);
            if (cc == 0) garr[M * 128 + t4] = p;
          }
        }
      }
      __syncthreads();

      // RK4 stage combine; emit packed-f16 zarg pairs for the next stage.
      for (int e2 = tid; e2 < 1024; e2 += 256) {
        int e = e2 * 2;
        float g0 = garr[e], g1 = garr[e + 1];
        float za0, za1;
        if (s == 0) {
          accl[e] = g0; accl[e + 1] = g1;
          za0 = zbase[e] + 0.5f * g0;
          za1 = zbase[e + 1] + 0.5f * g1;
        } else if (s == 1) {
          accl[e] += 2.0f * g0; accl[e + 1] += 2.0f * g1;
          za0 = zbase[e] + 0.5f * g0;
          za1 = zbase[e + 1] + 0.5f * g1;
        } else if (s == 2) {
          accl[e] += 2.0f * g0; accl[e + 1] += 2.0f * g1;
          za0 = zbase[e] + g0;
          za1 = zbase[e + 1] + g1;
        } else {
          za0 = zbase[e] + (accl[e] + g0) * (1.0f / 6.0f);
          za1 = zbase[e + 1] + (accl[e + 1] + g1) * (1.0f / 6.0f);
          zbase[e] = za0;
          zbase[e + 1] = za1;
          float2 zz; zz.x = za0; zz.y = za1;
          *(float2*)&zT[((long long)(bq + (e >> 7)) * 256 + (n + 1)) * 128 + (e & 127)] = zz;
        }
        zargU[e2] = pkh2(za0, za1);
      }
      __syncthreads();
    }
  }
}

// ---------------------------------------------------------------------------
// Generic WMMA GEMM over a strided view:
//   C[m, n] = act( A_view[m, :K] . W[n, :K] + bias[n] )
// token m: b = m>>sShift, s = m&(S-1); A elem = A[b*bS + s*rS + k*kS].
// Block = 128 threads (4 waves), tile = 16 rows x 64 cols, K staged in LDS f16
// as packed pairs; fragment feeds are pure ds_load_b128.
// ---------------------------------------------------------------------------
__global__ __launch_bounds__(128) void wmma_gemm_view(
    const float* __restrict__ A, long long bS, long long rS, long long kS,
    int sShift, const float* __restrict__ W, const float* __restrict__ bias,
    float* __restrict__ C, int kShift, int N, int act) {
  __shared__ unsigned AsmU[16 * 128];  // 16 x 256 halves max
  __shared__ unsigned BsmU[64 * 128];  // 64 x 256 halves max

  int K = 1 << kShift;
  int m0 = blockIdx.x * 16;
  int n0 = blockIdx.y * 64;
  int b = m0 >> sShift;
  int s0 = m0 & ((1 << sShift) - 1);
  const float* Ab = A + (long long)b * bS + (long long)s0 * rS;

  // stage A (pairs). kS==1 fast path uses contiguous float2 loads.
  if (kS == 1) {
    for (int e = threadIdx.x; e < (16 << kShift) / 2; e += 128) {
      int r = e >> (kShift - 1), kk = (e & ((K >> 1) - 1)) << 1;
      float2 a = *(const float2*)&Ab[(long long)r * rS + kk];
      AsmU[e] = pkh2(a.x, a.y);
    }
  } else {
    for (int e = threadIdx.x; e < (16 << kShift) / 2; e += 128) {
      int r = e >> (kShift - 1), kk = (e & ((K >> 1) - 1)) << 1;
      const float* ap = Ab + (long long)r * rS + (long long)kk * kS;
      AsmU[e] = pkh2(ap[0], ap[kS]);
    }
  }
  // stage B (W rows are contiguous)
  for (int e = threadIdx.x; e < (64 << kShift) / 2; e += 128) {
    int nn = e >> (kShift - 1), kk = (e & ((K >> 1) - 1)) << 1;
    float2 wv = *(const float2*)&W[(long long)(n0 + nn) * K + kk];
    BsmU[e] = pkh2(wv.x, wv.y);
  }
  __syncthreads();

  int w = threadIdx.x >> 5, lane = threadIdx.x & 31;
  const unsigned* Bw = BsmU + (((w * 16) << kShift) >> 1);
  v8f acc = {};
  for (int kf = 0; kf < (K >> 5); ++kf) {
    v16h a = packA_h16(AsmU, K, kf * 32, lane);
    v16h bb = packB_h16(Bw, K, kf * 32, lane);
    acc = wmma32(a, bb, acc);
  }

  int nn = n0 + w * 16 + (lane & 15);
  float bv = bias[nn];
#pragma unroll
  for (int r = 0; r < 8; ++r) {
    int m = m0 + r + ((lane >> 4) << 3);
    float v = acc[r] + bv;
    if (act == 1) v = 0.5f * v * (1.0f + erff(v * 0.70710678118654752f));
    C[(long long)m * N + nn] = v;
  }
}

// ---------------------------------------------------------------------------
// Attention: one block per (q-tile, head, batch). scores = softmax(QK^T * scale)
// then O = scores @ V. dh in {16,32}; K-dim zero-padded to 32 for dh=16.
// Probabilities are re-emitted as packed f16 so the P@V sweep reads b128 frags.
// ---------------------------------------------------------------------------
__global__ __launch_bounds__(128) void attn_kernel(
    const float* __restrict__ Q, const float* __restrict__ Kp,
    const float* __restrict__ V, float* __restrict__ O, int sShift, int D,
    int dh, float scale) {
  int S = 1 << sShift;
  int qt = blockIdx.x, hd = blockIdx.y, b = blockIdx.z;
  long long tb = (long long)b << sShift;
  __shared__ float sc[16 * 256];
  __shared__ unsigned scHU[16 * 128];  // 16 x 256 halves
  __shared__ float part[2048];
  int tid = threadIdx.x;
  int w = tid >> 5, lane = tid & 31;

  // Q fragment (A): row = q row, K-dim = head dim (zero padded)
  v16h qa;
  {
    union { v16h v; unsigned u[8]; } r;
    int qrow = qt * 16 + (lane & 15);
    const float* qp = Q + (tb + qrow) * D + hd * dh;
    int hw = (lane >> 4) << 4;
#pragma unroll
    for (int j = 0; j < 8; ++j) {
      int d0 = hw + (j << 1);
      float f0 = (d0 < dh) ? qp[d0] : 0.0f;
      float f1 = (d0 + 1 < dh) ? qp[d0 + 1] : 0.0f;
      r.u[j] = pkh2(f0, f1);
    }
    qa = r.v;
  }

  // scores
  for (int ct = w; ct < (S >> 4); ct += 4) {
    union { v16h v; unsigned u[8]; } kb;
    int key = ct * 16 + (lane & 15);
    const float* kp = Kp + (tb + key) * D + hd * dh;
    int hw = (lane >> 4) << 4;
#pragma unroll
    for (int j = 0; j < 8; ++j) {
      int d0 = hw + (j << 1);
      float f0 = (d0 < dh) ? kp[d0] : 0.0f;
      float f1 = (d0 + 1 < dh) ? kp[d0 + 1] : 0.0f;
      kb.u[j] = pkh2(f0, f1);
    }
    v8f a = {};
    a = wmma32(qa, kb.v, a);
#pragma unroll
    for (int r = 0; r < 8; ++r) {
      int m = r + ((lane >> 4) << 3);
      sc[m * S + ct * 16 + (lane & 15)] = a[r] * scale;
    }
  }
  __syncthreads();

  // row softmax (16 rows), emitting packed-f16 probabilities
  if (tid < 16) {
    float* row = sc + tid * S;
    unsigned* rowH = scHU + ((tid * S) >> 1);
    float mx = -3.0e38f;
    for (int j = 0; j < S; ++j) mx = fmaxf(mx, row[j]);
    float sum = 0.0f;
    for (int j = 0; j < S; ++j) {
      float e = expf(row[j] - mx);
      row[j] = e;
      sum += e;
    }
    float inv = 1.0f / sum;
    for (int j2 = 0; j2 < (S >> 1); ++j2)
      rowH[j2] = pkh2(row[2 * j2] * inv, row[2 * j2 + 1] * inv);
  }
  __syncthreads();

  // O = P @ V ; split K-dim (keys) across waves, reduce partials via LDS.
  int OT = dh >> 4;  // 1 or 2 output tiles
  v8f oacc[2] = {{}, {}};
  for (int kf = w; kf < (S >> 5); kf += 4) {
    v16h pa = packA_h16(scHU, S, kf * 32, lane);
#pragma unroll
    for (int o = 0; o < 2; ++o) {
      if (o >= OT) break;
      union { v16h v; unsigned u[8]; } vb;
      int dcol = hd * dh + o * 16 + (lane & 15);
      int keyb = kf * 32 + ((lane >> 4) << 4);
#pragma unroll
      for (int j = 0; j < 8; ++j) {
        int key0 = keyb + (j << 1);
        vb.u[j] = pkh2(V[(tb + key0) * D + dcol], V[(tb + key0 + 1) * D + dcol]);
      }
      oacc[o] = wmma32(pa, vb.v, oacc[o]);
    }
  }
#pragma unroll
  for (int o = 0; o < 2; ++o) {
    if (o >= OT) break;
#pragma unroll
    for (int r = 0; r < 8; ++r)
      part[((w * 2 + o) * 8 + r) * 32 + lane] = oacc[o][r];
  }
  __syncthreads();
  if (tid < 32) {
    int l2 = tid;
    for (int o = 0; o < OT; ++o) {
#pragma unroll
      for (int r = 0; r < 8; ++r) {
        float sres = part[((0 * 2 + o) * 8 + r) * 32 + l2] +
                     part[((1 * 2 + o) * 8 + r) * 32 + l2] +
                     part[((2 * 2 + o) * 8 + r) * 32 + l2] +
                     part[((3 * 2 + o) * 8 + r) * 32 + l2];
        int m = qt * 16 + r + ((l2 >> 4) << 3);
        int col = hd * dh + o * 16 + (l2 & 15);
        O[(tb + m) * D + col] = sres;
      }
    }
  }
}

// ---------------------------------------------------------------------------
// LayerNorm with residual: out[m,:] = LN(X_view[m,:] + R[m,:]) * gamma + beta
// ---------------------------------------------------------------------------
__global__ __launch_bounds__(128) void ln_kernel(
    const float* __restrict__ X, long long bS, long long rS, long long kS,
    int sShift, const float* __restrict__ R, const float* __restrict__ gamma,
    const float* __restrict__ beta, float* __restrict__ out, int D) {
  int m = blockIdx.x;
  int b = m >> sShift;
  int s = m & ((1 << sShift) - 1);
  const float* xb = X + (long long)b * bS + (long long)s * rS;
  const float* rb = R + (long long)m * D;
  int cnt = D >> 7;  // 1 or 2
  float v[2] = {0.0f, 0.0f};
  float sum = 0.0f, sq = 0.0f;
  for (int i = 0; i < cnt; ++i) {
    int d = threadIdx.x + i * 128;
    float val = xb[(long long)d * kS] + rb[d];
    v[i] = val;
    sum += val;
    sq += val * val;
  }
  __shared__ float red[128], red2[128];
  red[threadIdx.x] = sum;
  red2[threadIdx.x] = sq;
  __syncthreads();
  for (int off = 64; off > 0; off >>= 1) {
    if (threadIdx.x < off) {
      red[threadIdx.x] += red[threadIdx.x + off];
      red2[threadIdx.x] += red2[threadIdx.x + off];
    }
    __syncthreads();
  }
  float mean = red[0] / (float)D;
  float var = red2[0] / (float)D - mean * mean;
  float rstd = rsqrtf(var + 1e-5f);
  for (int i = 0; i < cnt; ++i) {
    int d = threadIdx.x + i * 128;
    out[(long long)m * D + d] = (v[i] - mean) * rstd * gamma[d] + beta[d];
  }
}

// ---------------------------------------------------------------------------
// Mean-pool (fts contiguous (b,l,h); fstE contiguous (b,h,l)) + classifier.
// ---------------------------------------------------------------------------
__global__ __launch_bounds__(128) void pool_clf_kernel(
    const float* __restrict__ fts, const float* __restrict__ fstE,
    const float* __restrict__ clfW, const float* __restrict__ clfb,
    float* __restrict__ out) {
  int b = blockIdx.x;
  int h = threadIdx.x;
  __shared__ float pool[256];
  float s1 = 0.0f, s2 = 0.0f;
  for (int l = 0; l < 256; ++l) s1 += fts[(long long)b * 32768 + l * 128 + h];
  for (int l = 0; l < 256; ++l) s2 += fstE[(long long)b * 32768 + h * 256 + l];
  pool[h] = s1 * (1.0f / 256.0f);
  pool[128 + h] = s2 * (1.0f / 256.0f);
  __syncthreads();
  if (h < 3) {
    float acc = clfb[h];
    for (int i = 0; i < 256; ++i) acc += clfW[h * 256 + i] * pool[i];
    out[b * 3 + h] = acc;
  }
}

// ---------------------------------------------------------------------------
// Host-side encoder orchestration
// ---------------------------------------------------------------------------
struct TransP {
  const float *Wq, *bq, *Wk, *bk, *Wv, *bv, *Wo, *bo;
  const float *ff1W, *ff1b, *ff2W, *ff2b;
  const float *ln1s, *ln1b, *ln2s, *ln2b;
};

static void run_enc(const float* X, long long bS, long long rS, long long kS,
                    int sShift, int D, int nh, const TransP& tp, float* Qb,
                    float* Kb, float* Vb, float* Ab, float* outBuf,
                    hipStream_t stream) {
  int S = 1 << sShift;
  long long M = 256LL * S;
  int kShiftD = (D == 256) ? 8 : 7;
  dim3 g1((unsigned)(M / 16), (unsigned)(D / 64));
  // Q, K, V projections
  wmma_gemm_view<<<g1, 128, 0, stream>>>(X, bS, rS, kS, sShift, tp.Wq, tp.bq, Qb, kShiftD, D, 0);
  wmma_gemm_view<<<g1, 128, 0, stream>>>(X, bS, rS, kS, sShift, tp.Wk, tp.bk, Kb, kShiftD, D, 0);
  wmma_gemm_view<<<g1, 128, 0, stream>>>(X, bS, rS, kS, sShift, tp.Wv, tp.bv, Vb, kShiftD, D, 0);
  int dh = D / nh;
  attn_kernel<<<dim3((unsigned)(S / 16), (unsigned)nh, 256u), 128, 0, stream>>>(
      Qb, Kb, Vb, Ab, sShift, D, dh, 1.0f / sqrtf((float)dh));
  // output projection (att contiguous) -> Qb
  wmma_gemm_view<<<g1, 128, 0, stream>>>(Ab, (long long)S * D, D, 1, sShift, tp.Wo, tp.bo, Qb, kShiftD, D, 0);
  // x1 = LN(X + O) -> Kb
  ln_kernel<<<(unsigned)M, 128, 0, stream>>>(X, bS, rS, kS, sShift, Qb, tp.ln1s, tp.ln1b, Kb, D);
  // F1 = gelu(x1 @ ff1^T + b) -> Vb (M x 64)
  wmma_gemm_view<<<dim3((unsigned)(M / 16), 1u), 128, 0, stream>>>(
      Kb, (long long)S * D, D, 1, sShift, tp.ff1W, tp.ff1b, Vb, kShiftD, 64, 1);
  // F2 = F1 @ ff2^T + b -> Qb (M x D)
  wmma_gemm_view<<<g1, 128, 0, stream>>>(Vb, (long long)S * 64, 64, 1, sShift, tp.ff2W, tp.ff2b, Qb, 6, D, 0);
  // out = LN(x1 + F2)
  ln_kernel<<<(unsigned)M, 128, 0, stream>>>(Kb, (long long)S * D, D, 1, sShift, Qb, tp.ln2s, tp.ln2b, outBuf, D);
}

// ---------------------------------------------------------------------------
// kernel_launch
//
// Input order assumes jax pytree (sorted-key) flattening of setup_inputs():
//   0: coeffs (256*255*64)
//   1: cde.W0  2: cde.W4  3: cde.Wm  4: cde.b0  5: cde.b4  6: cde.bm
//   7: clf_W   8: clf_b   9: init_W  10: init_b
//   11..26: s params (Wk,Wo,Wq,Wv,bk,bo,bq,bv,ff1_W,ff1_b,ff2_W,ff2_b,
//                     ln1_b,ln1_s,ln2_b,ln2_s)
//   27..42: t params (same key order)
//   43: x (unused by the forward pass)
// ---------------------------------------------------------------------------
extern "C" void kernel_launch(void* const* d_in, const int* in_sizes, int n_in,
                              void* d_out, int out_size, void* d_ws,
                              size_t ws_size, hipStream_t stream) {
  (void)in_sizes; (void)n_in; (void)out_size; (void)ws_size;
  const float* coeffs = (const float*)d_in[0];
  const float* cdeW0 = (const float*)d_in[1];
  const float* cdeW4 = (const float*)d_in[2];
  const float* cdeWm = (const float*)d_in[3];
  const float* cdeb0 = (const float*)d_in[4];
  const float* cdeb4 = (const float*)d_in[5];
  const float* cdebm = (const float*)d_in[6];
  const float* clfW = (const float*)d_in[7];
  const float* clfb = (const float*)d_in[8];
  const float* initW = (const float*)d_in[9];
  const float* initb = (const float*)d_in[10];

  auto loadTP = [&](int base) {
    TransP t;
    t.Wk = (const float*)d_in[base + 0];
    t.Wo = (const float*)d_in[base + 1];
    t.Wq = (const float*)d_in[base + 2];
    t.Wv = (const float*)d_in[base + 3];
    t.bk = (const float*)d_in[base + 4];
    t.bo = (const float*)d_in[base + 5];
    t.bq = (const float*)d_in[base + 6];
    t.bv = (const float*)d_in[base + 7];
    t.ff1W = (const float*)d_in[base + 8];
    t.ff1b = (const float*)d_in[base + 9];
    t.ff2W = (const float*)d_in[base + 10];
    t.ff2b = (const float*)d_in[base + 11];
    t.ln1b = (const float*)d_in[base + 12];
    t.ln1s = (const float*)d_in[base + 13];
    t.ln2b = (const float*)d_in[base + 14];
    t.ln2s = (const float*)d_in[base + 15];
    return t;
  };
  TransP tpS = loadTP(11);
  TransP tpT = loadTP(27);

  // workspace carve (floats, 64B aligned chunks)
  float* ws = (float*)d_ws;
  size_t off = 0;
  auto take = [&](size_t nElems) {
    float* p = ws + off;
    off += (nElems + 15) & ~(size_t)15;
    return p;
  };
  const size_t MD = 8388608;  // 256*256*128 == 256*128*256
  float* zT = take(MD);
  float* ft = take(MD);
  float* fts = take(MD);
  float* fs = take(MD);
  float* fstE = take(MD);
  float* Qb = take(MD);
  float* Kb = take(MD);
  float* Vb = take(MD);
  float* Ab = take(MD);
  unsigned* pW0 = (unsigned*)take(4096);
  unsigned* pWm = (unsigned*)take(4096);
  unsigned* pW4 = (unsigned*)take(65536);

  // --- pack ODE weights into WMMA B-fragment order ---
  cde_pack_kernel<<<(4096 + 255) / 256, 256, 0, stream>>>(cdeW0, pW0, 128, 4);
  cde_pack_kernel<<<(2048 + 255) / 256, 256, 0, stream>>>(cdeWm, pWm, 64, 4);
  cde_pack_kernel<<<(2048 + 255) / 256, 256, 0, stream>>>(cdeWm + 64 * 64, pWm + 2048, 64, 4);
  cde_pack_kernel<<<(65536 + 255) / 256, 256, 0, stream>>>(cdeW4, pW4, 64, 128);

  // --- spline + z0 ---
  spline_z0_kernel<<<256, 128, 0, stream>>>(coeffs, initW, initb, zT);

  // --- persistent RK4 integration -> zT[:, 1:255, :] ---
  cde_integrate_kernel<<<16, 256, 0, stream>>>(coeffs, pW0, pWm, pW4, cdeb0,
                                               cdebm, cdeb4, zT);

  // --- transformer encoders (strided views, no materialized transposes) ---
  // enc1: ft = enc(zT^T(0,2,1)) : tokens (b, d in [0,128)), model k over L=256
  run_enc(zT, 32768, 1, 128, 7, 256, 8, tpT, Qb, Kb, Vb, Ab, ft, stream);
  // enc2: fts = enc(ft^T) : tokens (b, l in [0,256)), model k over 128
  run_enc(ft, 32768, 1, 256, 8, 128, 8, tpS, Qb, Kb, Vb, Ab, fts, stream);
  // enc3: fs = enc(zT) : contiguous (b, l, h)
  run_enc(zT, 32768, 128, 1, 8, 128, 8, tpS, Qb, Kb, Vb, Ab, fs, stream);
  // enc4: fstE = enc(fs^T) : tokens (b, h in [0,128)), model k over L=256
  run_enc(fs, 32768, 1, 128, 7, 256, 8, tpT, Qb, Kb, Vb, Ab, fstE, stream);

  // --- pooling + classifier ---
  pool_clf_kernel<<<256, 128, 0, stream>>>(fts, fstE, clfW, clfb,
                                           (float*)d_out);
}
